// BMOJOFAttentionLayer_16690242912527
// MI455X (gfx1250) — compile-verified, matrix operand order checked by reading, not run on hardware
//
#include <hip/hip_runtime.h>

typedef __attribute__((ext_vector_type(16))) _Float16 v16h;
typedef __attribute__((ext_vector_type(2)))  _Float16 v2h;
typedef __attribute__((ext_vector_type(8)))  float    v8f;

#define S_LEN  2048
#define NH     16
#define HD     128
#define WINDOW 512      // union of the two reference streams: 0 <= i-j < 512
#define QW     16       // queries per wave (one WMMA M-tile)
#define WAVES  4
#define QB     (QW*WAVES)
#define KT     32       // key tile (one WMMA K=32 for PV)
#define KPAD   8
#define VPAD   4
#define PPAD   4

__global__ __launch_bounds__(128)
void bmojof_swa_wmma(const float* __restrict__ Q,
                     const float* __restrict__ K,
                     const float* __restrict__ V,
                     float* __restrict__ O)
{
    __shared__ _Float16 sK [KT][HD + KPAD];        // K tile, [key][dim], f16
    __shared__ _Float16 sVT[HD][KT + VPAD];        // V tile transposed, [dim][key], f16
    __shared__ _Float16 sP [WAVES][QW][KT + PPAD]; // per-wave P tile, [m][key], f16

    const int b     = blockIdx.z;
    const int h     = blockIdx.y;
    const int qbase = blockIdx.x * QB;
    const int tid   = threadIdx.x;
    const int wave  = tid >> 5;
    const int lane  = tid & 31;
    const int lm    = lane & 15;    // M (A/row) or N (B/C col)
    const int hh    = lane >> 4;    // lane-half selector
    const int qw    = qbase + wave * QW;
    const float scale = 0.08838834764831845f; // 128^-0.5

    // ---- Load this wave's Q tile straight into WMMA A-operand layout ----
    // A 16-bit 16x32: lane%16 = M; elem e -> Kdim = (e<8 ? e : e+8) + 8*(lane/16)
    const float* qrow = Q + (((size_t)b * S_LEN + (qw + lm)) * NH + h) * HD;
    v16h aq[4];
    #pragma unroll
    for (int c = 0; c < 4; ++c) {
        #pragma unroll
        for (int p = 0; p < 8; ++p) {
            int kd = 32 * c + 2 * p + (p >= 4 ? 8 : 0) + 8 * hh;
            float2 t = *(const float2*)(qrow + kd);
            aq[c][2 * p]     = (_Float16)t.x;
            aq[c][2 * p + 1] = (_Float16)t.y;
        }
    }

    // ---- Online-softmax state (rows M = c + 8*hh of this wave's tile) ----
    float m[8], l[8];
    v8f   acc[8]; // D=128 -> 8 x (16x16 f32) accumulators
    #pragma unroll
    for (int c = 0; c < 8; ++c) { m[c] = -1e30f; l[c] = 0.0f; }
    #pragma unroll
    for (int dk = 0; dk < 8; ++dk) {
        #pragma unroll
        for (int e = 0; e < 8; ++e) acc[dk][e] = 0.0f;
    }

    int ks = qbase - (WINDOW - 1);
    if (ks < 0) ks = 0;
    ks &= ~(KT - 1);
    const int ke = qbase + QB;

    for (int kt = ks; kt < ke; kt += KT) {
        // ---- Cooperative stage of K (row-major) and V (transposed) as f16 ----
        {
            const int r  = tid >> 2;        // key row within tile 0..31
            const int ch = (tid & 3) * 32;  // 32-dim chunk
            const size_t rowoff = (((size_t)b * S_LEN + (kt + r)) * NH + h) * HD + ch;
            const float* krow = K + rowoff;
            const float* vrow = V + rowoff;
            if (kt + KT < ke) { // hint next tile into cache (global_prefetch_b8)
                __builtin_prefetch(krow + (size_t)KT * NH * HD, 0, 1);
                __builtin_prefetch(vrow + (size_t)KT * NH * HD, 0, 1);
            }
            #pragma unroll
            for (int d4 = 0; d4 < 32; d4 += 4) {
                float4 kk = *(const float4*)(krow + d4);
                float4 vv = *(const float4*)(vrow + d4);
                sK[r][ch + d4 + 0] = (_Float16)kk.x;
                sK[r][ch + d4 + 1] = (_Float16)kk.y;
                sK[r][ch + d4 + 2] = (_Float16)kk.z;
                sK[r][ch + d4 + 3] = (_Float16)kk.w;
                sVT[ch + d4 + 0][r] = (_Float16)vv.x;
                sVT[ch + d4 + 1][r] = (_Float16)vv.y;
                sVT[ch + d4 + 2][r] = (_Float16)vv.z;
                sVT[ch + d4 + 3][r] = (_Float16)vv.w;
            }
        }
        __syncthreads();

        // wave-uniform guard: does this tile intersect this wave's window?
        const bool active = (kt + KT - 1 >= qw - (WINDOW - 1)) && (kt <= qw + QW - 1);
        if (active) {
            // ---- S = Q K^T over two 16-key subtiles, K=32 chunks of head-dim ----
            v8f s0, s1;
            #pragma unroll
            for (int e = 0; e < 8; ++e) { s0[e] = 0.0f; s1[e] = 0.0f; }
            #pragma unroll
            for (int c = 0; c < 4; ++c) {
                v16h bk0, bk1;
                #pragma unroll
                for (int p = 0; p < 8; ++p) {
                    int kd = 32 * c + 2 * p + (p >= 4 ? 8 : 0) + 8 * hh;
                    v2h t0 = *(const v2h*)&sK[lm][kd];
                    v2h t1 = *(const v2h*)&sK[16 + lm][kd];
                    bk0[2 * p] = t0.x; bk0[2 * p + 1] = t0.y;
                    bk1[2 * p] = t1.x; bk1[2 * p + 1] = t1.y;
                }
                s0 = __builtin_amdgcn_wmma_f32_16x16x32_f16(false, aq[c], false, bk0,
                                                            (short)0, s0, false, false);
                s1 = __builtin_amdgcn_wmma_f32_16x16x32_f16(false, aq[c], false, bk1,
                                                            (short)0, s1, false, false);
            }

            // ---- mask + online softmax (rows M = c + 8*hh) ----
            float alpha[8];
            #pragma unroll
            for (int c = 0; c < 8; ++c) {
                const int i  = qw + c + 8 * hh;
                const int j0 = kt + lm;
                const int j1 = kt + 16 + lm;
                float x0 = (j0 <= i && j0 > i - WINDOW) ? s0[c] * scale : -1e30f;
                float x1 = (j1 <= i && j1 > i - WINDOW) ? s1[c] * scale : -1e30f;
                float mt = fmaxf(x0, x1);
                mt = fmaxf(mt, __shfl_xor(mt, 1, 32));
                mt = fmaxf(mt, __shfl_xor(mt, 2, 32));
                mt = fmaxf(mt, __shfl_xor(mt, 4, 32));
                mt = fmaxf(mt, __shfl_xor(mt, 8, 32));
                float mn = fmaxf(m[c], mt);
                float a  = __expf(m[c] - mn);
                m[c] = mn;
                float p0 = __expf(x0 - mn);
                float p1 = __expf(x1 - mn);
                sP[wave][c + 8 * hh][lm]      = (_Float16)p0;
                sP[wave][c + 8 * hh][16 + lm] = (_Float16)p1;
                float rs = p0 + p1;
                rs += __shfl_xor(rs, 1, 32);
                rs += __shfl_xor(rs, 2, 32);
                rs += __shfl_xor(rs, 4, 32);
                rs += __shfl_xor(rs, 8, 32);
                l[c] = l[c] * a + rs;
                alpha[c] = a;
            }
            v8f av;
            #pragma unroll
            for (int c = 0; c < 8; ++c) av[c] = alpha[c];
            #pragma unroll
            for (int dk = 0; dk < 8; ++dk) acc[dk] *= av;

            // ---- P back from LDS in A-operand layout (16x32, K=keys) ----
            v16h pa;
            #pragma unroll
            for (int p = 0; p < 8; ++p) {
                int kk2 = 2 * p + (p >= 4 ? 8 : 0) + 8 * hh;
                v2h t = *(const v2h*)&sP[wave][lm][kk2];
                pa[2 * p] = t.x; pa[2 * p + 1] = t.y;
            }
            // ---- O += P V : B-operand from transposed V tile ----
            #pragma unroll
            for (int dk = 0; dk < 8; ++dk) {
                v16h vb;
                #pragma unroll
                for (int p = 0; p < 8; ++p) {
                    int kk2 = 2 * p + (p >= 4 ? 8 : 0) + 8 * hh;
                    v2h t = *(const v2h*)&sVT[dk * 16 + lm][kk2];
                    vb[2 * p] = t.x; vb[2 * p + 1] = t.y;
                }
                acc[dk] = __builtin_amdgcn_wmma_f32_16x16x32_f16(false, pa, false, vb,
                                                                 (short)0, acc[dk], false, false);
            }
        }
        __syncthreads();
    }

    // ---- normalize and store fp32 output ----
    #pragma unroll
    for (int c = 0; c < 8; ++c) {
        const int i = qw + c + 8 * hh;
        const float inv = 1.0f / l[c];
        float* orow = O + (((size_t)b * S_LEN + i) * NH + h) * HD;
        #pragma unroll
        for (int dk = 0; dk < 8; ++dk)
            orow[dk * 16 + lm] = acc[dk][c] * inv;
    }
}

extern "C" void kernel_launch(void* const* d_in, const int* in_sizes, int n_in,
                              void* d_out, int out_size, void* d_ws, size_t ws_size,
                              hipStream_t stream) {
    (void)n_in; (void)d_ws; (void)ws_size; (void)out_size;
    const float* q = (const float*)d_in[0];
    const float* k = (const float*)d_in[1];
    const float* v = (const float*)d_in[2];
    float* out = (float*)d_out;
    const int B = in_sizes[0] / (S_LEN * NH * HD);
    dim3 grid(S_LEN / QB, NH, B);
    dim3 block(128, 1, 1);
    bmojof_swa_wmma<<<grid, block, 0, stream>>>(q, k, v, out);
}
